// ConvMoE_51711406244008
// MI455X (gfx1250) — matrix operand order repaired and checked by reading, not compiled, not run on hardware
//
#include <hip/hip_runtime.h>
#include <hip/hip_bf16.h>
#include <cstdint>
#include <cstddef>

// Problem constants
#define NTOK   25088     // B*H*W = 32*28*28
#define CCH    384
#define HIDD   384
#define NEXP   4
#define HWSZ   784       // 28*28
#define TILE_M 64
#define XS     392       // padded bf16 row stride (384 + 8) -> conflict-free ds_load_b128

typedef __attribute__((ext_vector_type(16))) __bf16 v16bf;
typedef __attribute__((ext_vector_type(8)))  float  v8f;

union FragU { uint4 u[2]; v16bf v; };
union B8U   { uint4 u;    __bf16 h[8]; };

// ---------------------------------------------------------------------------
// Kernel 1: convert w1, w2 to bf16; zero the present-bitmask word
// ---------------------------------------------------------------------------
__global__ __launch_bounds__(256)
void prep_weights(const float* __restrict__ w1, const float* __restrict__ w2,
                  __bf16* __restrict__ w1b, __bf16* __restrict__ w2b,
                  unsigned* __restrict__ present)
{
    int idx = blockIdx.x * 256 + threadIdx.x;
    if (idx == 0) *present = 0u;
    const int n = NEXP * HIDD * CCH;   // 589824 = 2304 * 256
    if (idx < n) {
        w1b[idx] = (__bf16)w1[idx];
        w2b[idx] = (__bf16)w2[idx];
    }
}

// ---------------------------------------------------------------------------
// Kernel 2: x NCHW fp32 -> xt [token][channel] bf16 (token = b*784 + hw)
// grid (25, 12, 32), block (32, 8)
// ---------------------------------------------------------------------------
__global__ __launch_bounds__(256)
void transpose_x(const float* __restrict__ x, __bf16* __restrict__ xt)
{
    __shared__ float tile[32][33];
    const int b   = blockIdx.z;
    const int hw0 = blockIdx.x * 32;
    const int c0  = blockIdx.y * 32;
    const int tx  = threadIdx.x;
    const int ty  = threadIdx.y;

    #pragma unroll
    for (int i = ty; i < 32; i += 8) {
        int hw = hw0 + tx;
        int c  = c0 + i;
        tile[i][tx] = (hw < HWSZ) ? x[((size_t)(b * CCH + c)) * HWSZ + hw] : 0.f;
    }
    __syncthreads();
    #pragma unroll
    for (int i = ty; i < 32; i += 8) {
        int hw = hw0 + i;
        int c  = c0 + tx;
        if (hw < HWSZ)
            xt[(size_t)(b * HWSZ + hw) * CCH + c] = (__bf16)tile[tx][i];
    }
}

// ---------------------------------------------------------------------------
// Kernel 3: gating. scores = softmax(x@gw + gb); top-2; softmax(top2) weights.
// Writes tw[0..N) = slot0 weight, tw[N..2N) = slot1 weight.
// OR-reduces present bitmask (bit k*4+e) per block, one atomicOr per block.
// grid 98 x 256 (= 25088 threads, one per token)
// ---------------------------------------------------------------------------
__global__ __launch_bounds__(256)
void gate_kernel(const __bf16* __restrict__ xt,
                 const float* __restrict__ gate_w, const float* __restrict__ gate_b,
                 float* __restrict__ tw, unsigned* __restrict__ present)
{
    __shared__ float gw[CCH * NEXP];   // 6 KB
    __shared__ unsigned pmask;
    const int tid = threadIdx.x;
    if (tid == 0) pmask = 0u;
    for (int i = tid; i < CCH * NEXP; i += 256) gw[i] = gate_w[i];
    __syncthreads();

    const int t = blockIdx.x * 256 + tid;
    float s0 = gate_b[0], s1 = gate_b[1], s2 = gate_b[2], s3 = gate_b[3];

    const uint4* row = (const uint4*)(xt + (size_t)t * CCH);
    #pragma unroll 4
    for (int ci = 0; ci < CCH / 8; ++ci) {
        B8U u; u.u = row[ci];
        #pragma unroll
        for (int j = 0; j < 8; ++j) {
            float xv = (float)u.h[j];
            int c = ci * 8 + j;
            s0 += xv * gw[c * 4 + 0];
            s1 += xv * gw[c * 4 + 1];
            s2 += xv * gw[c * 4 + 2];
            s3 += xv * gw[c * 4 + 3];
        }
    }
    // softmax over 4
    float p[4] = {s0, s1, s2, s3};
    float m = fmaxf(fmaxf(p[0], p[1]), fmaxf(p[2], p[3]));
    float sum = 0.f;
    #pragma unroll
    for (int e = 0; e < 4; ++e) { p[e] = __expf(p[e] - m); sum += p[e]; }
    float inv = 1.f / sum;
    #pragma unroll
    for (int e = 0; e < 4; ++e) p[e] *= inv;

    // top-2 (ties -> lower index, matching top_k)
    int i1 = 0; float p1 = p[0];
    #pragma unroll
    for (int e = 1; e < 4; ++e) if (p[e] > p1) { p1 = p[e]; i1 = e; }
    int i2 = (i1 == 0) ? 1 : 0; float p2 = p[i2];
    #pragma unroll
    for (int e = 0; e < 4; ++e) if (e != i1 && p[e] > p2) { p2 = p[e]; i2 = e; }

    // softmax over the two selected probabilities (p1 >= p2)
    float e1 = 1.f;                 // exp(p1 - p1)
    float e2 = __expf(p2 - p1);
    float rs = 1.f / (e1 + e2);
    tw[t]        = e1 * rs;
    tw[NTOK + t] = e2 * rs;

    atomicOr(&pmask, (1u << i1) | (1u << (4 + i2)));
    __syncthreads();
    if (tid == 0) atomicOr(present, pmask);
}

// ---------------------------------------------------------------------------
// Kernel 4: fused dense MoE MLP with per-token gated accumulation.
// grid 392 x 256 (8 waves). Each WG: 64 tokens x full 384 output channels.
// Wave layout: mg = wave&3 -> 16-token strip, ng = wave>>2 -> 192-col strip.
// n-tiles processed in blocks of 6 inside the k loop: the A fragment is
// loaded once per k-step (2 ds_load_b128) and feeds 6 WMMAs whose B frags
// are clause-loaded from one base pointer with immediate offsets.
// ---------------------------------------------------------------------------
#define NTB 6                 // n-tiles per block
#define BSTEP (16 * CCH / 8)  // uint4 stride between adjacent n-tiles' B rows

__global__ __launch_bounds__(256)
void moe_main(const __bf16* __restrict__ xt,
              const __bf16* __restrict__ w1b,
              const __bf16* __restrict__ w2b,
              const float* __restrict__ b1,
              const float* __restrict__ b2,
              const float* __restrict__ tw,
              const unsigned* __restrict__ presentp,
              float* __restrict__ out)
{
    extern __shared__ char smem[];
    __bf16* Xs = (__bf16*)smem;                 // [64][392] bf16
    __bf16* Hs = Xs + TILE_M * XS;              // [64][392] bf16
    float*  Ts = (float*)Hs;                    // overlay (used only after final barrier)

    const int tid     = threadIdx.x;
    const int wave    = tid >> 5;
    const int lane    = tid & 31;
    const int lanecol = lane & 15;
    const int khalf   = lane >> 4;              // 0 or 1
    const int wm0     = (wave & 3) * 16;
    const int wn0     = (wave >> 2) * 192;
    const int t0      = blockIdx.x * TILE_M;

    // ---- stage X tile (64 rows x 48 x 16B chunks), coalesced ----
    {
        const uint4* src = (const uint4*)(xt + (size_t)t0 * CCH);
        for (int ch = tid; ch < TILE_M * (CCH / 8); ch += 256) {
            int row = ch / (CCH / 8);
            int cc  = ch % (CCH / 8);
            uint4 v = src[(size_t)row * (CCH / 8) + cc];
            *((uint4*)(Xs + row * XS) + cc) = v;
        }
    }

    const unsigned P = *presentp;

    // per-lane-half row data: rows rbase..rbase+7 of the C/D fragment
    const int rbase = wm0 + (khalf << 3);
    float tw0v[8], tw1v[8];
    {
        const float4* p0 = (const float4*)(tw + t0 + rbase);
        const float4* p1 = (const float4*)(tw + NTOK + t0 + rbase);
        float4 a0 = p0[0], a1 = p0[1];
        float4 c0 = p1[0], c1 = p1[1];
        tw0v[0]=a0.x; tw0v[1]=a0.y; tw0v[2]=a0.z; tw0v[3]=a0.w;
        tw0v[4]=a1.x; tw0v[5]=a1.y; tw0v[6]=a1.z; tw0v[7]=a1.w;
        tw1v[0]=c0.x; tw1v[1]=c0.y; tw1v[2]=c0.z; tw1v[3]=c0.w;
        tw1v[4]=c1.x; tw1v[5]=c1.y; tw1v[6]=c1.z; tw1v[7]=c1.w;
    }

    v8f oacc[12];
    #pragma unroll
    for (int nt = 0; nt < 12; ++nt)
        #pragma unroll
        for (int i = 0; i < 8; ++i) oacc[nt][i] = 0.f;

    __syncthreads();

    for (int e = 0; e < NEXP; ++e) {
        const __bf16* w1e = w1b + (size_t)e * HIDD * CCH;
        const __bf16* w2e = w2b + (size_t)e * CCH * HIDD;

        // ---- GEMM1: H = relu(X @ W1^T + b1), bf16 into LDS ----
        #pragma unroll
        for (int ntb = 0; ntb < 2; ++ntb) {
            const int colb = wn0 + ntb * (NTB * 16) + lanecol;
            v8f acc[NTB];
            #pragma unroll
            for (int j = 0; j < NTB; ++j) {
                float bias = b1[e * HIDD + colb + j * 16];
                #pragma unroll
                for (int i = 0; i < 8; ++i) acc[j][i] = bias;
            }
            #pragma unroll 2
            for (int k = 0; k < 12; ++k) {
                const int kb = k * 32;
                FragU a;
                const uint4* ap = (const uint4*)(Xs + (wm0 + lanecol) * XS + kb + khalf * 8);
                a.u[0] = ap[0];          // K = kb+kh*8 .. +7
                a.u[1] = ap[2];          // K = kb+kh*8+16 .. +23
                const uint4* bp = (const uint4*)(w1e + (size_t)colb * CCH + kb + khalf * 8);
                #pragma unroll
                for (int j = 0; j < NTB; ++j) {
                    FragU b;
                    b.u[0] = bp[j * BSTEP];       // K = kb+kh*8 .. +7, then kb+kh*8+16 .. +23
                    b.u[1] = bp[j * BSTEP + 2];
                    acc[j] = __builtin_amdgcn_wmma_f32_16x16x32_bf16(
                                 false, a.v, false, b.v, (short)0, acc[j], false, false);
                }
            }
            #pragma unroll
            for (int j = 0; j < NTB; ++j)
                #pragma unroll
                for (int i = 0; i < 8; ++i) {
                    float h = acc[j][i] > 0.f ? acc[j][i] : 0.f;
                    Hs[(rbase + i) * XS + colb + j * 16] = (__bf16)h;
                }
        }
        __syncthreads();

        // per-row gate coefficient for this expert
        const float pe0 = (float)((P >> e) & 1u);
        const float pe1 = (float)((P >> (4 + e)) & 1u);
        float coefr[8];
        #pragma unroll
        for (int i = 0; i < 8; ++i) coefr[i] = pe0 * tw0v[i] + pe1 * tw1v[i];

        // ---- GEMM2: Y = H @ W2^T + b2; out += coef * Y ----
        #pragma unroll
        for (int ntb = 0; ntb < 2; ++ntb) {
            const int colb = wn0 + ntb * (NTB * 16) + lanecol;
            v8f acc[NTB];
            #pragma unroll
            for (int j = 0; j < NTB; ++j) {
                float bias = b2[e * CCH + colb + j * 16];
                #pragma unroll
                for (int i = 0; i < 8; ++i) acc[j][i] = bias;
            }
            #pragma unroll 2
            for (int k = 0; k < 12; ++k) {
                const int kb = k * 32;
                FragU a;
                const uint4* ap = (const uint4*)(Hs + (wm0 + lanecol) * XS + kb + khalf * 8);
                a.u[0] = ap[0];
                a.u[1] = ap[2];
                const uint4* bp = (const uint4*)(w2e + (size_t)colb * HIDD + kb + khalf * 8);
                #pragma unroll
                for (int j = 0; j < NTB; ++j) {
                    FragU b;
                    b.u[0] = bp[j * BSTEP];
                    b.u[1] = bp[j * BSTEP + 2];
                    acc[j] = __builtin_amdgcn_wmma_f32_16x16x32_bf16(
                                 false, a.v, false, b.v, (short)0, acc[j], false, false);
                }
            }
            #pragma unroll
            for (int j = 0; j < NTB; ++j)
                #pragma unroll
                for (int i = 0; i < 8; ++i)
                    oacc[ntb * NTB + j][i] += coefr[i] * acc[j][i];
        }
        __syncthreads();
    }

    // ---- coalesced NCHW store via per-wave LDS transpose (overlaid on Hs) ----
    float* T = Ts + wave * (16 * 17);
    const int tokl = lanecol;
    const int tmy  = t0 + wm0 + tokl;
    const int bb   = tmy / HWSZ;
    const int hwv  = tmy % HWSZ;
    #pragma unroll 1
    for (int nt = 0; nt < 12; ++nt) {
        #pragma unroll
        for (int i = 0; i < 8; ++i)
            T[((khalf << 3) + i) * 17 + lanecol] = oacc[nt][i];
        asm volatile("s_wait_dscnt 0" ::: "memory");
        #pragma unroll
        for (int j = 0; j < 8; ++j) {
            int lcol = 2 * j + khalf;
            float val = T[tokl * 17 + lcol];
            int gcol = wn0 + nt * 16 + lcol;
            out[((size_t)(bb * CCH + gcol)) * HWSZ + hwv] = val;
        }
        asm volatile("s_wait_dscnt 0" ::: "memory");
    }
}

// ---------------------------------------------------------------------------
extern "C" void kernel_launch(void* const* d_in, const int* in_sizes, int n_in,
                              void* d_out, int out_size, void* d_ws, size_t ws_size,
                              hipStream_t stream)
{
    (void)in_sizes; (void)n_in; (void)out_size; (void)ws_size;
    const float* x      = (const float*)d_in[0];
    const float* gate_w = (const float*)d_in[1];
    const float* gate_b = (const float*)d_in[2];
    const float* w1     = (const float*)d_in[3];
    const float* b1     = (const float*)d_in[4];
    const float* w2     = (const float*)d_in[5];
    const float* b2     = (const float*)d_in[6];
    float* out = (float*)d_out;

    char* ws = (char*)d_ws;
    size_t off = 0;
    __bf16* xt  = (__bf16*)(ws + off); off += (size_t)NTOK * CCH * 2;        // 19,267,584
    __bf16* w1b = (__bf16*)(ws + off); off += (size_t)NEXP * HIDD * CCH * 2; //  1,179,648
    __bf16* w2b = (__bf16*)(ws + off); off += (size_t)NEXP * CCH * HIDD * 2; //  1,179,648
    float*  twp = (float*)(ws + off);  off += (size_t)2 * NTOK * 4;          //    200,704
    unsigned* presentp = (unsigned*)(ws + off);                              // 4 bytes

    prep_weights<<<2304, 256, 0, stream>>>(w1, w2, w1b, w2b, presentp);

    dim3 tgrid(25, 12, 32), tblk(32, 8);
    transpose_x<<<tgrid, tblk, 0, stream>>>(x, xt);

    gate_kernel<<<NTOK / 256, 256, 0, stream>>>(xt, gate_w, gate_b, twp, presentp);

    const size_t lds_bytes = (size_t)2 * TILE_M * XS * 2;   // Xs + Hs, 100,352 B
    moe_main<<<NTOK / TILE_M, 256, lds_bytes, stream>>>(
        xt, w1b, w2b, b1, b2, twp, presentp, out);
}
